// GQN_77000173683416
// MI455X (gfx1250) — matrix operand order repaired and checked by reading, not compiled, not run on hardware
//
#include <hip/hip_runtime.h>
#include <hip/hip_bf16.h>
#include <math.h>

typedef __attribute__((ext_vector_type(16))) _Float16 v16h;
typedef __attribute__((ext_vector_type(8)))  float    v8f;

#define NN   50176
#define EE   802816
#define GG   64
#define HID  128
#define GLOB 16

// ---------------------------------------------------------------------------
// Operand pre-swizzle into a WMMA fragment layout (contiguous k per lane half), f32 -> f16.
// A fragment (16x32):  a[i] = A[mt*16 + (lane&15)][ks*32 + (lane>>4)*16 + i]
// B fragment (32x16):  b[i] = W[ks*32 + (lane>>4)*16 + i][nt*16 + (lane&15)]
// Stored so each fragment is 16 contiguous f16 (32B) per lane.
// ---------------------------------------------------------------------------
__global__ void swizzleA(const float* __restrict__ A, _Float16* __restrict__ Ah,
                         int M)
{
    int t = blockIdx.x * blockDim.x + threadIdx.x;     // (M/16)*4*32 threads
    if (t >= (M / 16) * 128) return;
    const int mt   = t >> 7;
    const int ks   = (t >> 5) & 3;
    const int lane = t & 31;
    const int half = lane >> 4, mr = lane & 15;
    const float* ap = A + (size_t)(mt * 16 + mr) * HID + ks * 32;
    _Float16* out = Ah + (size_t)t * 16;
    #pragma unroll
    for (int i = 0; i < 8; ++i)  out[i]     = (_Float16)ap[half * 16 + i];
    #pragma unroll
    for (int i = 0; i < 8; ++i)  out[8 + i] = (_Float16)ap[half * 16 + 8 + i];
}

__global__ void swizzleW(const float* __restrict__ W, _Float16* __restrict__ Wh)
{
    int t = blockIdx.x * blockDim.x + threadIdx.x;     // 4*8*32 = 1024 threads
    if (t >= 1024) return;
    const int ks   = t >> 8;
    const int nt   = (t >> 5) & 7;
    const int lane = t & 31;
    const int half = lane >> 4;
    const int col  = nt * 16 + (lane & 15);
    _Float16* out = Wh + (size_t)t * 16;
    #pragma unroll
    for (int i = 0; i < 16; ++i)
        out[i] = (_Float16)W[(size_t)(ks * 32 + half * 16 + i) * HID + col];
}

// ---------------------------------------------------------------------------
// WMMA GEMM on pre-swizzled operands:
//   C[M,128] = A[M,128] @ W[128,128] (+bias) (+gvec[gid[row]]) (+relu)
// One wave computes a full 16x128 row-block: 8 fp32 accumulators, A fragment
// reused 8x. All 8 B fragments of a k-step are preloaded (one load clause)
// so the 8 WMMAs issue back-to-back with partial loadcnt waits.
// ---------------------------------------------------------------------------
__global__ void wmma_gemm128(const _Float16* __restrict__ Ah,
                             const _Float16* __restrict__ Wh,
                             const float* __restrict__ bias,
                             float* __restrict__ C, int M, int fuse_relu,
                             const float* __restrict__ gvec,
                             const int* __restrict__ gid)
{
    const int lane = threadIdx.x & 31;
    const int mt = blockIdx.x * (blockDim.x >> 5) + (threadIdx.x >> 5);
    if (mt * 16 >= M) return;          // wave-uniform

    v8f acc[8] = {};
    #pragma unroll
    for (int ks = 0; ks < 4; ++ks) {
        const v16h a = *(const v16h*)(Ah + ((size_t)(mt * 4 + ks) * 32 + lane) * 16);
        v16h b[8];
        #pragma unroll
        for (int nt = 0; nt < 8; ++nt)
            b[nt] = *(const v16h*)(Wh + ((size_t)(ks * 8 + nt) * 32 + lane) * 16);
        #pragma unroll
        for (int nt = 0; nt < 8; ++nt)
            acc[nt] = __builtin_amdgcn_wmma_f32_16x16x32_f16(
                false, a, false, b[nt], (short)0, acc[nt], false, false);
    }

    // D layout: VGPR r -> row r (lanes 0-15) / r+8 (lanes 16-31)
    const int half = lane >> 4;
    if (gvec) {
        #pragma unroll
        for (int nt = 0; nt < 8; ++nt) {
            const int ncol = nt * 16 + (lane & 15);
            const float bcol = bias[ncol];
            #pragma unroll
            for (int r = 0; r < 8; ++r) {
                int row = mt * 16 + r + half * 8;
                float v = acc[nt][r] + bcol + gvec[gid[row] * HID + ncol];
                if (fuse_relu) v = fmaxf(v, 0.0f);
                C[(size_t)row * HID + ncol] = v;
            }
        }
    } else {
        #pragma unroll
        for (int nt = 0; nt < 8; ++nt) {
            const int ncol = nt * 16 + (lane & 15);
            const float bcol = bias[ncol];
            #pragma unroll
            for (int r = 0; r < 8; ++r) {
                int row = mt * 16 + r + half * 8;
                float v = acc[nt][r] + bcol;
                if (fuse_relu) v = fmaxf(v, 0.0f);
                C[(size_t)row * HID + ncol] = v;
            }
        }
    }
}

// ---------------------------------------------------------------------------
// Per-layer init: segment-max = -inf, segment-sum = 0, BN stats = 0
// ---------------------------------------------------------------------------
__global__ void init_layer(float* m, float* ssum, float* stats, int M)
{
    int i = blockIdx.x * blockDim.x + threadIdx.x;
    if (i < M) { m[i] = -__builtin_inff(); ssum[i] = 0.0f; }
    if (i < 256) stats[i] = 0.0f;
}

// ---------------------------------------------------------------------------
// Edge pass 1: logits[e] = q[dst]·k[src] * scale ; atomic segment-max on m
// Wave-per-edge: 32 lanes x float4 = 128-wide coalesced row gather
// ---------------------------------------------------------------------------
__global__ void edge_logits_max(const int* __restrict__ src,
                                const int* __restrict__ dst,
                                const float* __restrict__ q,
                                const float* __restrict__ k,
                                float* __restrict__ logits,
                                float* __restrict__ m, int nE)
{
    const int lane = threadIdx.x & 31;
    const int e = blockIdx.x * (blockDim.x >> 5) + (threadIdx.x >> 5);
    if (e >= nE) return;
    const int d = dst[e], s = src[e];
    const float4 qv = ((const float4*)(q + (size_t)d * HID))[lane];
    const float4 kv = ((const float4*)(k + (size_t)s * HID))[lane];
    float p = qv.x * kv.x + qv.y * kv.y + qv.z * kv.z + qv.w * kv.w;
    #pragma unroll
    for (int off = 16; off >= 1; off >>= 1) p += __shfl_xor(p, off, 32);
    p *= 0.0883883476483184f;   // 1/sqrt(128)
    if (lane == 0) {
        logits[e] = p;
        // IEEE-ordered atomic max: signed-int max for >=0, unsigned min for <0
        if (p >= 0.0f) atomicMax((int*)(m + d), __float_as_int(p));
        else           atomicMin((unsigned int*)(m + d), __float_as_uint(p));
    }
}

// ---------------------------------------------------------------------------
// Edge pass 2: e = exp(logit - m[dst]); segment-sum via fp32 atomic add
// ---------------------------------------------------------------------------
__global__ void edge_exp_sum(const int* __restrict__ dst,
                             const float* __restrict__ m,
                             float* __restrict__ logits,
                             float* __restrict__ ssum, int nE)
{
    int e = blockIdx.x * blockDim.x + threadIdx.x;
    if (e >= nE) return;
    int d = dst[e];
    float ex = expf(logits[e] - m[d]);
    logits[e] = ex;                      // reuse buffer for exp value
    unsafeAtomicAdd(ssum + d, ex);       // global_atomic_add_f32
}

// ---------------------------------------------------------------------------
// Edge pass 3: agg[dst] += alpha * v[src]   (agg pre-loaded with skip branch)
// ---------------------------------------------------------------------------
__global__ void edge_aggregate(const int* __restrict__ src,
                               const int* __restrict__ dst,
                               const float* __restrict__ v,
                               const float* __restrict__ eexp,
                               const float* __restrict__ ssum,
                               float* __restrict__ agg, int nE)
{
    const int lane = threadIdx.x & 31;
    const int e = blockIdx.x * (blockDim.x >> 5) + (threadIdx.x >> 5);
    if (e >= nE) return;
    const int d = dst[e], s = src[e];
    const float alpha = eexp[e] / (ssum[d] + 1e-16f);
    const float4 vv = ((const float4*)(v + (size_t)s * HID))[lane];
    float* ap = agg + (size_t)d * HID + lane * 4;
    unsafeAtomicAdd(ap + 0, alpha * vv.x);
    unsafeAtomicAdd(ap + 1, alpha * vv.y);
    unsafeAtomicAdd(ap + 2, alpha * vv.z);
    unsafeAtomicAdd(ap + 3, alpha * vv.w);
}

// ---------------------------------------------------------------------------
// BatchNorm: column sum / sumsq partials -> atomics; finalize; apply+ReLU
// ---------------------------------------------------------------------------
__global__ void bn_partial(const float* __restrict__ t, float* stats, int M)
{
    const int col = threadIdx.x;   // 128 threads
    float s = 0.0f, q = 0.0f;
    for (int r = blockIdx.x; r < M; r += gridDim.x) {
        float v = t[(size_t)r * HID + col];
        s += v; q += v * v;
    }
    unsafeAtomicAdd(stats + col, s);
    unsafeAtomicAdd(stats + HID + col, q);
}

__global__ void bn_finalize(const float* __restrict__ stats,
                            const float* __restrict__ gamma,
                            const float* __restrict__ beta,
                            float* __restrict__ ab, int M)
{
    int c = threadIdx.x;
    float mu  = stats[c] / (float)M;
    float var = stats[HID + c] / (float)M - mu * mu;
    float inv = rsqrtf(var + 1e-5f);
    float a = gamma[c] * inv;
    ab[c] = a;
    ab[HID + c] = beta[c] - a * mu;
}

__global__ void bn_apply_relu(const float* __restrict__ t,
                              const float* __restrict__ ab,
                              float* __restrict__ h, int M)
{
    size_t i = (size_t)blockIdx.x * blockDim.x + threadIdx.x;
    if (i >= (size_t)M * HID) return;
    int c = (int)(i & (HID - 1));
    h[i] = fmaxf(ab[c] * t[i] + ab[HID + c], 0.0f);
}

// ---------------------------------------------------------------------------
// graph id per node (searchsorted(ptr, i, 'right') - 1)
// ---------------------------------------------------------------------------
__global__ void compute_gid(const int* __restrict__ ptr, int* __restrict__ gid, int M)
{
    int i = blockIdx.x * blockDim.x + threadIdx.x;
    if (i >= M) return;
    int g = 0;
    #pragma unroll 1
    for (int j = 1; j <= GG; ++j) { if (ptr[j] <= i) g = j; else break; }
    gid[i] = g;
}

// gc[g][j] = sum_k global_features[g][k] * fcW[128+k][j]
__global__ void gcontrib_kernel(const float* __restrict__ gf,
                                const float* __restrict__ fcW,
                                float* __restrict__ gc)
{
    int t = blockIdx.x * blockDim.x + threadIdx.x;
    if (t >= GG * HID) return;
    int g = t >> 7, j = t & (HID - 1);
    float s = 0.0f;
    #pragma unroll
    for (int k = 0; k < GLOB; ++k)
        s += gf[g * GLOB + k] * fcW[(HID + k) * HID + j];
    gc[t] = s;
}

// out[i] = fc_h[i,:] . fc2W + fc2b   (wave-per-node)
__global__ void fc2_kernel(const float* __restrict__ h,
                           const float* __restrict__ w,
                           const float* __restrict__ b,
                           float* __restrict__ out, int M)
{
    const int lane = threadIdx.x & 31;
    const int i = blockIdx.x * (blockDim.x >> 5) + (threadIdx.x >> 5);
    if (i >= M) return;
    const float4 hv = ((const float4*)(h + (size_t)i * HID))[lane];
    const float4 wv = ((const float4*)w)[lane];
    float p = hv.x * wv.x + hv.y * wv.y + hv.z * wv.z + hv.w * wv.w;
    #pragma unroll
    for (int off = 16; off >= 1; off >>= 1) p += __shfl_xor(p, off, 32);
    if (lane == 0) out[i] = p + b[0];
}

// ---------------------------------------------------------------------------
// host-side layer driver
// ---------------------------------------------------------------------------
static void run_tconv_layer(const float* in, const int* src, const int* dst,
                            const _Float16* WhQ, const _Float16* WhK,
                            const _Float16* WhV, const _Float16* WhS,
                            const float* bq, const float* bk,
                            const float* bv, const float* bs,
                            const float* gamma, const float* beta,
                            _Float16* Ah,
                            float* bufQ, float* bufK, float* bufV, float* bufA,
                            float* bufE, float* bufM, float* bufS,
                            float* stats, float* ab, float* bufH,
                            hipStream_t stream)
{
    const int gemmBlocks = (NN / 16) / 8;           // 1 wave per 16-row block
    const int swzBlocks  = ((NN / 16) * 128) / 256;
    hipLaunchKernelGGL(init_layer, dim3((NN + 255) / 256), dim3(256), 0, stream,
                       bufM, bufS, stats, NN);
    hipLaunchKernelGGL(swizzleA, dim3(swzBlocks), dim3(256), 0, stream, in, Ah, NN);
    hipLaunchKernelGGL(wmma_gemm128, dim3(gemmBlocks), dim3(256), 0, stream,
                       Ah, WhQ, bq, bufQ, NN, 0, (const float*)nullptr, (const int*)nullptr);
    hipLaunchKernelGGL(wmma_gemm128, dim3(gemmBlocks), dim3(256), 0, stream,
                       Ah, WhK, bk, bufK, NN, 0, (const float*)nullptr, (const int*)nullptr);
    hipLaunchKernelGGL(wmma_gemm128, dim3(gemmBlocks), dim3(256), 0, stream,
                       Ah, WhV, bv, bufV, NN, 0, (const float*)nullptr, (const int*)nullptr);
    hipLaunchKernelGGL(wmma_gemm128, dim3(gemmBlocks), dim3(256), 0, stream,
                       Ah, WhS, bs, bufA, NN, 0, (const float*)nullptr, (const int*)nullptr);
    hipLaunchKernelGGL(edge_logits_max, dim3(EE / 8), dim3(256), 0, stream,
                       src, dst, bufQ, bufK, bufE, bufM, EE);
    hipLaunchKernelGGL(edge_exp_sum, dim3(EE / 256), dim3(256), 0, stream,
                       dst, bufM, bufE, bufS, EE);
    hipLaunchKernelGGL(edge_aggregate, dim3(EE / 8), dim3(256), 0, stream,
                       src, dst, bufV, bufE, bufS, bufA, EE);
    hipLaunchKernelGGL(bn_partial, dim3(512), dim3(HID), 0, stream, bufA, stats, NN);
    hipLaunchKernelGGL(bn_finalize, dim3(1), dim3(HID), 0, stream, stats, gamma, beta, ab, NN);
    hipLaunchKernelGGL(bn_apply_relu, dim3((NN * HID) / 256), dim3(256), 0, stream,
                       bufA, ab, bufH, NN);
}

extern "C" void kernel_launch(void* const* d_in, const int* in_sizes, int n_in,
                              void* d_out, int out_size, void* d_ws, size_t ws_size,
                              hipStream_t stream)
{
    const float* x    = (const float*)d_in[0];
    const int*   eidx = (const int*)d_in[1];
    const int*   ptr  = (const int*)d_in[2];
    const float* gf   = (const float*)d_in[3];
    const float *Wq0 = (const float*)d_in[4],  *Wk0 = (const float*)d_in[5];
    const float *Wv0 = (const float*)d_in[6],  *Ws0 = (const float*)d_in[7];
    const float *bq0 = (const float*)d_in[8],  *bk0 = (const float*)d_in[9];
    const float *bv0 = (const float*)d_in[10], *bs0 = (const float*)d_in[11];
    const float *g0  = (const float*)d_in[12], *be0 = (const float*)d_in[13];
    const float *Wq1 = (const float*)d_in[14], *Wk1 = (const float*)d_in[15];
    const float *Wv1 = (const float*)d_in[16], *Ws1 = (const float*)d_in[17];
    const float *bq1 = (const float*)d_in[18], *bk1 = (const float*)d_in[19];
    const float *bv1 = (const float*)d_in[20], *bs1 = (const float*)d_in[21];
    const float *g1  = (const float*)d_in[22], *be1 = (const float*)d_in[23];
    const float *fcW = (const float*)d_in[24], *fcb = (const float*)d_in[25];
    const float *fc2W = (const float*)d_in[26], *fc2b = (const float*)d_in[27];

    const int* src = eidx;
    const int* dst = eidx + EE;

    // workspace layout (fp32-element granularity)
    float* ws = (float*)d_ws;
    size_t o = 0;
    float* bufA = ws + o; o += (size_t)NN * HID;   // skip + aggregation
    float* bufQ = ws + o; o += (size_t)NN * HID;
    float* bufK = ws + o; o += (size_t)NN * HID;
    float* bufV = ws + o; o += (size_t)NN * HID;
    float* bufH = ws + o; o += (size_t)NN * HID;   // layer output
    float* bufE = ws + o; o += (size_t)EE;         // logits -> exp
    float* bufM = ws + o; o += (size_t)NN;         // segment max
    float* bufS = ws + o; o += (size_t)NN;         // segment sum
    float* stats = ws + o; o += 256;               // col sum / sumsq
    float* ab    = ws + o; o += 256;               // fused BN scale/shift
    float* gc    = ws + o; o += (size_t)GG * HID;  // global-feature contrib
    int*   gid   = (int*)(ws + o); o += (size_t)NN;
    _Float16* Ah = (_Float16*)(ws + o); o += (size_t)NN * HID / 2;  // swizzled f16 activations
    _Float16* WhBase = (_Float16*)(ws + o); o += (9 * 16384) / 2;   // 9 swizzled f16 weights
    _Float16* Wh[9];
    for (int i = 0; i < 9; ++i) Wh[i] = WhBase + (size_t)i * 16384;

    // pre-swizzle all 9 weight matrices (cheap, deterministic per call)
    const float* Wsrc[9] = { Wq0, Wk0, Wv0, Ws0, Wq1, Wk1, Wv1, Ws1, fcW };
    for (int i = 0; i < 9; ++i)
        hipLaunchKernelGGL(swizzleW, dim3(4), dim3(256), 0, stream, Wsrc[i], Wh[i]);

    // layer 1 (input x) and layer 2 (input bufH)
    run_tconv_layer(x,    src, dst, Wh[0], Wh[1], Wh[2], Wh[3],
                    bq0, bk0, bv0, bs0, g0, be0, Ah,
                    bufQ, bufK, bufV, bufA, bufE, bufM, bufS,
                    stats, ab, bufH, stream);
    run_tconv_layer(bufH, src, dst, Wh[4], Wh[5], Wh[6], Wh[7],
                    bq1, bk1, bv1, bs1, g1, be1, Ah,
                    bufQ, bufK, bufV, bufA, bufE, bufM, bufS,
                    stats, ab, bufH, stream);

    // FC over [h || global_features[graph]] with fused relu, then scalar head
    hipLaunchKernelGGL(compute_gid, dim3((NN + 255) / 256), dim3(256), 0, stream,
                       ptr, gid, NN);
    hipLaunchKernelGGL(gcontrib_kernel, dim3((GG * HID + 255) / 256), dim3(256), 0, stream,
                       gf, fcW, gc);
    const int gemmBlocks = (NN / 16) / 8;
    const int swzBlocks  = ((NN / 16) * 128) / 256;
    hipLaunchKernelGGL(swizzleA, dim3(swzBlocks), dim3(256), 0, stream, bufH, Ah, NN);
    hipLaunchKernelGGL(wmma_gemm128, dim3(gemmBlocks), dim3(256), 0, stream,
                       Ah, Wh[8], fcb, bufQ, NN, 1, gc, gid);   // fc uses first 128 rows of fcW
    hipLaunchKernelGGL(fc2_kernel, dim3(NN / 8), dim3(256), 0, stream,
                       bufQ, fc2W, fc2b, (float*)d_out, NN);
}
